// SelfAttention_16655883174250
// MI455X (gfx1250) — matrix operand order repaired and checked by reading, not compiled, not run on hardware
//
#include <hip/hip_runtime.h>
#include <hip/hip_bf16.h>

// Problem constants (from the reference)
#define B_  2
#define N_  2048
#define D_  1024
#define H_  8
#define DK_ 128
#define DV_ 1024
#define INV_N (1.0f / 2048.0f)

typedef __attribute__((ext_vector_type(16))) __bf16 v16bf;
typedef __attribute__((ext_vector_type(8)))  __bf16 v8bf;
typedef __attribute__((ext_vector_type(8)))  float  v8f;

// ---------- helpers ----------

__device__ inline unsigned short f2bf(float f) {
  union { float f; unsigned u; } x; x.f = f;
  unsigned r = x.u + 0x7FFFu + ((x.u >> 16) & 1u);  // round-to-nearest-even
  return (unsigned short)(r >> 16);
}

__device__ inline v8f wmma_bf16(v16bf a, v16bf b, v8f c) {
  // D = A(16x32) * B(32x16) + C, f32 accumulate
  return __builtin_amdgcn_wmma_f32_16x16x32_bf16(
      /*neg_a=*/false, a, /*neg_b=*/false, b,
      /*c_mod=*/(short)0, c, /*reuse_a=*/false, /*reuse_b=*/false);
}

// A-operand (16x32, row-major source, leading dim ld in elements).
// Lane (m, half): elements 0-7 = K[8*half .. 8*half+7], 8-15 = K[16+8*half ..].
__device__ inline v16bf ld_a(const __bf16* base, int ld, int m, int half) {
  const __bf16* p = base + (size_t)m * (size_t)ld + (half << 3);
  v8bf lo = *(const v8bf*)(p);
  v8bf hi = *(const v8bf*)(p + 16);
  v16bf r;
#pragma unroll
  for (int i = 0; i < 8; ++i) { r[i] = lo[i]; r[8 + i] = hi[i]; }
  return r;
}

// B-operand (32x16): column n must be contiguous in memory (ld between columns).
// Lane (n, half): elements 0-7 = K[8*half .. 8*half+7], 8-15 = K[16+8*half ..].
__device__ inline v16bf ld_b(const __bf16* base, int ld, int n, int half) {
  const __bf16* p = base + (size_t)n * (size_t)ld + (half << 3);
  v8bf lo = *(const v8bf*)(p);
  v8bf hi = *(const v8bf*)(p + 16);
  v16bf r;
#pragma unroll
  for (int i = 0; i < 8; ++i) { r[i] = lo[i]; r[8 + i] = hi[i]; }
  return r;
}

// ---------- kernel 1: fp32 -> bf16 convert ----------

__global__ __launch_bounds__(256) void cvt_kernel(const float* __restrict__ in,
                                                  unsigned short* __restrict__ out,
                                                  int n4) {
  int i = blockIdx.x * blockDim.x + threadIdx.x;
  int stride = gridDim.x * blockDim.x;
  const float4* in4 = (const float4*)in;
  for (; i < n4; i += stride) {
    float4 v = in4[i];
    unsigned long long pk = (unsigned long long)f2bf(v.x)
                          | ((unsigned long long)f2bf(v.y) << 16)
                          | ((unsigned long long)f2bf(v.z) << 32)
                          | ((unsigned long long)f2bf(v.w) << 48);
    *(unsigned long long*)(out + 4 * (size_t)i) = pk;
  }
}

// ---------- kernel 2: projection GEMM ----------
// Out(row, col) = sum_d X[row, d] * W[col, d];  X: [4096, D], W: [OutCols, D] row-major.
// mode 0: out[B,H,N,hd] (Q/K, hd=DK);  mode 1: out[B,H,hd,N] transposed (V, hd=DV).
// Each wave computes a 32x64 output tile.
__global__ __launch_bounds__(256) void proj_kernel(const unsigned short* __restrict__ Xus,
                                                   const unsigned short* __restrict__ Wus,
                                                   unsigned short* __restrict__ Ous,
                                                   int ctShift, int hShift, int mode) {
  const __bf16* X = (const __bf16*)Xus;
  const __bf16* W = (const __bf16*)Wus;
  int lane = threadIdx.x & 31;
  int w = threadIdx.x >> 5;
  int m = lane & 15, half = lane >> 4;
  int tile = blockIdx.x * 8 + w;
  int rt = tile >> ctShift;
  int cc = tile & ((1 << ctShift) - 1);
  int rg0 = rt * 32, c0 = cc * 64;

  v8f acc[2][4] = {};
  const __bf16* a0base = X + (size_t)rg0 * D_;

  for (int k = 0; k < D_; k += 32) {
    v16bf a0 = ld_a(a0base + k, D_, m, half);
    v16bf a1 = ld_a(a0base + (size_t)16 * D_ + k, D_, m, half);
#pragma unroll
    for (int j = 0; j < 4; ++j) {
      v16bf bb = ld_b(W + (size_t)(c0 + 16 * j) * D_ + k, D_, m, half);
      acc[0][j] = wmma_bf16(a0, bb, acc[0][j]);
      acc[1][j] = wmma_bf16(a1, bb, acc[1][j]);
    }
  }

#pragma unroll
  for (int i = 0; i < 2; ++i) {
#pragma unroll
    for (int j = 0; j < 4; ++j) {
#pragma unroll
      for (int r = 0; r < 8; ++r) {
        int row  = rg0 + 16 * i + r + (half << 3);
        int colg = c0 + 16 * j + m;
        int b = row >> 11, n = row & (N_ - 1);
        int h = colg >> hShift;
        int e = colg & ((1 << hShift) - 1);
        size_t addr;
        if (mode == 0) addr = (((size_t)(b * H_ + h) * N_ + n) << hShift) + e;
        else           addr = ((((size_t)(b * H_ + h) << hShift) + e) * N_) + n;
        Ous[addr] = f2bf(acc[i][j][r]);
      }
    }
  }
}

// ---------- kernel 3: fused causal-relu attention ----------
// Block = (b, 32-row tile, DV half of 512). 8 waves / 256 threads.
// Stage 1: each wave computes one 16x16 score subtile of the 32x64 sim tile,
//          applies causal+relu+1/N, stores bf16 into double-buffered LDS.
//          Q A-operands are hoisted into registers once per head.
// Stage 2: each wave owns 64 DV columns: sim(32x64) @ Vt(64x64), acc in VGPRs,
//          summed over all heads and key tiles.
__global__ __launch_bounds__(256) void attn_kernel(const unsigned short* __restrict__ Qus,
                                                   const unsigned short* __restrict__ Kus,
                                                   const unsigned short* __restrict__ Vus,
                                                   const float* __restrict__ x,
                                                   float* __restrict__ out) {
  __shared__ __align__(16) unsigned short simlds[2][32 * 72];  // ld=72 pads banks

  int lane = threadIdx.x & 31;
  int w = threadIdx.x >> 5;
  int m = lane & 15, half = lane >> 4;
  int bid = blockIdx.x;
  int dvblk = bid & 1;
  int rblk = (bid >> 1) & 63;
  int b = bid >> 7;
  int r0 = rblk * 32;
  int mt_max = (r0 + 32 + 63) >> 6;   // key tiles needed under causal mask
  int sr = w >> 2, sc = w & 3;        // stage-1 subtile coords
  int dv0 = dvblk * 512 + w * 64;     // stage-2 DV columns for this wave

  v8f acc[2][4] = {};

  for (int h = 0; h < H_; ++h) {
    const __bf16* Qh = (const __bf16*)Qus + (size_t)(b * H_ + h) * N_ * DK_;
    const __bf16* Kh = (const __bf16*)Kus + (size_t)(b * H_ + h) * N_ * DK_;
    const __bf16* Vh = (const __bf16*)Vus + (size_t)(b * H_ + h) * DV_ * (size_t)N_;

    // Hoist this wave's Q rows (16 x DK) into registers for the whole key loop.
    v16bf qa[4];
    {
      const __bf16* qbase = Qh + (size_t)(r0 + 16 * sr) * DK_;
#pragma unroll
      for (int k4 = 0; k4 < 4; ++k4) qa[k4] = ld_a(qbase + 32 * k4, DK_, m, half);
    }

    for (int mt = 0; mt < mt_max; ++mt) {
      int m0 = mt * 64;
      int buf = mt & 1;

      // ---- stage 1: scores = Q . K^T (16x16 per wave, K-dim = DK) ----
      v8f s = {};
      const __bf16* kbase = Kh + (size_t)(m0 + 16 * sc) * DK_;
#pragma unroll
      for (int k4 = 0; k4 < 4; ++k4) {
        v16bf bb = ld_b(kbase + 32 * k4, DK_, m, half);  // B col n = K row (contiguous)
        s = wmma_bf16(qa[k4], bb, s);
      }

      // Prefetch next key tile's K rows and Vt rows (global_prefetch_b8).
      if (mt + 1 < mt_max) {
        __builtin_prefetch(Kh + (size_t)(m0 + 64 + 16 * sc + m) * DK_, 0, 3);
        __builtin_prefetch(Vh + (size_t)(dv0 + (lane << 1)) * N_ + m0 + 64, 0, 3);
      }

      int colj = m0 + 16 * sc + m;
#pragma unroll
      for (int r = 0; r < 8; ++r) {
        int rowi = r0 + 16 * sr + r + (half << 3);
        float v = (colj <= rowi) ? fmaxf(s[r], 0.0f) * INV_N : 0.0f;
        simlds[buf][(16 * sr + r + (half << 3)) * 72 + 16 * sc + m] = f2bf(v);
      }
      __syncthreads();

      // ---- stage 2: acc += sim(32x64) @ Vt(64 x 64dv) ----
#pragma unroll
      for (int kk = 0; kk < 2; ++kk) {
        const __bf16* sbase = (const __bf16*)&simlds[buf][kk * 32];
        v16bf a0 = ld_a(sbase, 72, m, half);
        v16bf a1 = ld_a(sbase + 16 * 72, 72, m, half);
#pragma unroll
        for (int j = 0; j < 4; ++j) {
          // B col n = dv column; Vt row (dv) is contiguous over keys
          v16bf bb = ld_b(Vh + (size_t)(dv0 + 16 * j) * N_ + m0 + kk * 32, N_, m, half);
          acc[0][j] = wmma_bf16(a0, bb, acc[0][j]);
          acc[1][j] = wmma_bf16(a1, bb, acc[1][j]);
        }
      }
      __syncthreads();
    }
  }

  // ---- epilogue: residual add, fp32 out ----
#pragma unroll
  for (int i = 0; i < 2; ++i) {
#pragma unroll
    for (int j = 0; j < 4; ++j) {
#pragma unroll
      for (int r = 0; r < 8; ++r) {
        int row = r0 + 16 * i + r + (half << 3);
        int col = dv0 + 16 * j + m;
        size_t idx = ((size_t)b * N_ + row) * DV_ + col;
        out[idx] = x[idx] + acc[i][j][r];
      }
    }
  }
}

// ---------- host launcher ----------

extern "C" void kernel_launch(void* const* d_in, const int* in_sizes, int n_in,
                              void* d_out, int out_size, void* d_ws, size_t ws_size,
                              hipStream_t stream) {
  (void)in_sizes; (void)n_in; (void)out_size;
  const float* x  = (const float*)d_in[0];
  const float* Wq = (const float*)d_in[1];
  const float* Wk = (const float*)d_in[2];
  const float* Wv = (const float*)d_in[3];
  float* out = (float*)d_out;

  char* ws = (char*)d_ws;
  size_t off = 0;
  unsigned short* xb  = (unsigned short*)(ws + off); off += (size_t)B_ * N_ * D_ * 2;        // 8 MB
  unsigned short* Wqb = (unsigned short*)(ws + off); off += (size_t)H_ * DK_ * D_ * 2;       // 2 MB
  unsigned short* Wkb = (unsigned short*)(ws + off); off += (size_t)H_ * DK_ * D_ * 2;       // 2 MB
  unsigned short* Wvb = (unsigned short*)(ws + off); off += (size_t)H_ * DV_ * D_ * 2;       // 16 MB
  unsigned short* Qb  = (unsigned short*)(ws + off); off += (size_t)B_ * H_ * N_ * DK_ * 2;  // 8 MB
  unsigned short* Kb  = (unsigned short*)(ws + off); off += (size_t)B_ * H_ * N_ * DK_ * 2;  // 8 MB
  unsigned short* Vtb = (unsigned short*)(ws + off); off += (size_t)B_ * H_ * DV_ * N_ * 2;  // 64 MB
  if (ws_size < off) return;  // workspace too small; nothing sane to do

  // fp32 -> bf16
  cvt_kernel<<<1024, 256, 0, stream>>>(x,  xb,  B_ * N_ * D_ / 4);
  cvt_kernel<<<256,  256, 0, stream>>>(Wq, Wqb, H_ * DK_ * D_ / 4);
  cvt_kernel<<<256,  256, 0, stream>>>(Wk, Wkb, H_ * DK_ * D_ / 4);
  cvt_kernel<<<1024, 256, 0, stream>>>(Wv, Wvb, H_ * DV_ * D_ / 4);

  // Q = x Wq^T, K = x Wk^T   -> [B,H,N,DK]   (OutCols=1024, colTiles=16 -> ctShift=4)
  proj_kernel<<<256,  256, 0, stream>>>(xb, Wqb, Qb, 4, 7, 0);
  proj_kernel<<<256,  256, 0, stream>>>(xb, Wkb, Kb, 4, 7, 0);
  // Vt = (x Wv^T)^T          -> [B,H,DV,N]   (OutCols=8192, colTiles=128 -> ctShift=7)
  proj_kernel<<<2048, 256, 0, stream>>>(xb, Wvb, Vtb, 7, 10, 1);

  // fused causal-relu attention + residual
  attn_kernel<<<B_ * (N_ / 32) * (DV_ / 512), 256, 0, stream>>>(Qb, Kb, Vtb, x, out);
}
